// GraphNeighborhoodReg_21414706938295
// MI455X (gfx1250) — matrix operand order repaired and checked by reading, not compiled, not run on hardware
//
#include <hip/hip_runtime.h>
#include <math.h>

#define NROWS 4096
#define DIM   64
#define KNN   15
#define NNEG  32
#define TEMPR 0.1f
#define MARGIN 0.5f
#define NEGINF (-1.0e30f)

typedef __attribute__((ext_vector_type(16))) _Float16 v16h;
typedef __attribute__((ext_vector_type(8)))  _Float16 v8h;
typedef __attribute__((ext_vector_type(8)))  float    v8f;

// ---- CDNA5 async global->LDS copy (ASYNCcnt-tracked), 16 bytes per lane ----
// GLOBAL_LOAD_ASYNC_TO_LDS_B128: VDST = LDS byte address (VGPR), VADDR = 64-bit
// global address (VGPR pair).  Followed later by s_wait_asynccnt 0 + barrier.
__device__ __forceinline__ void async_ld_lds_b128(float* lds_dst, const float* gsrc) {
  unsigned laddr =
      (unsigned)(unsigned long long)(__attribute__((address_space(3))) float*)lds_dst;
  unsigned long long gaddr = (unsigned long long)gsrc;
  asm volatile("global_load_async_to_lds_b128 %0, %1, off"
               :
               : "v"(laddr), "v"(gaddr)
               : "memory");
}
__device__ __forceinline__ void wait_asynccnt0() {
  asm volatile("s_wait_asynccnt 0x0" ::: "memory");
}

// ---------------------------------------------------------------- init
__global__ void k_init(float* acc) {
  if (threadIdx.x < 8) acc[threadIdx.x] = 0.0f;
}

// ---------------------------------------------------------------- row-normalize + f16 convert
// rows [0,N)   : z_rna  -> zrh (f16)
// rows [N,2N)  : z_atac -> zan (f32) + zah (f16)
__global__ void __launch_bounds__(256) k_normalize(const float* __restrict__ zr,
                                                   const float* __restrict__ za,
                                                   _Float16* __restrict__ zrh,
                                                   _Float16* __restrict__ zah,
                                                   float* __restrict__ zan) {
  const int lane = threadIdx.x & 31;
  const int warp = threadIdx.x >> 5;
  const int row  = blockIdx.x * 8 + warp;            // 0 .. 2N-1
  const bool atac = row >= NROWS;
  const int r = atac ? row - NROWS : row;
  const float* src = atac ? za : zr;
  float2 v = ((const float2*)(src + (size_t)r * DIM))[lane];
  float s = v.x * v.x + v.y * v.y;
  #pragma unroll
  for (int off = 16; off > 0; off >>= 1) s += __shfl_down(s, off, 32);
  s = __shfl(s, 0, 32);
  const float inv = 1.0f / fmaxf(sqrtf(s), 1e-12f);
  const float x = v.x * inv, y = v.y * inv;
  if (atac) {
    ((float2*)(zan + (size_t)r * DIM))[lane] = make_float2(x, y);
    zah[(size_t)r * DIM + 2 * lane]     = (_Float16)x;
    zah[(size_t)r * DIM + 2 * lane + 1] = (_Float16)y;
  } else {
    zrh[(size_t)r * DIM + 2 * lane]     = (_Float16)x;
    zrh[(size_t)r * DIM + 2 * lane + 1] = (_Float16)y;
  }
}

// ---------------------------------------------------------------- WMMA similarity GEMM: S = Z * Z^T
// A fragment (16x32 f16): lane L<16 holds row M=L, K in {kb+0..15};
//                         lane L>=16 holds row M=L-16, K in {kb+16..31}.
__device__ inline v16h load_afrag(const _Float16* __restrict__ Z, int row, int koff) {
  const _Float16* p = Z + (size_t)row * DIM + koff;
  v8h lo = *(const v8h*)(p);
  v8h hi = *(const v8h*)(p + 8);
  return __builtin_shufflevector(lo, hi, 0, 1, 2, 3, 4, 5, 6, 7,
                                         8, 9, 10, 11, 12, 13, 14, 15);
}

__global__ void __launch_bounds__(256) k_sim_gemm(const _Float16* __restrict__ Z,
                                                  float* __restrict__ S) {
  const int lane = threadIdx.x & 31;
  const int wave = threadIdx.x >> 5;
  const int i0 = blockIdx.y * 128 + (wave >> 1) * 32;   // output rows
  const int j0 = blockIdx.x * 64 + (wave & 1) * 32;     // output cols
  const int ar = i0 + (lane & 15);
  const int ak = (lane >> 4) * 16;                      // A K sub-offset
  const int bc = j0 + (lane & 15);                      // B column (= Z row)
  const int bk = (lane >> 4) * 16;                      // B K sub-offset

  v8f c00 = {}, c01 = {}, c10 = {}, c11 = {};
  #pragma unroll
  for (int kb = 0; kb < DIM; kb += 32) {
    v16h a0 = load_afrag(Z, ar,      kb + ak);
    v16h a1 = load_afrag(Z, ar + 16, kb + ak);
    v16h b0 = *(const v16h*)(Z + (size_t)bc * DIM + kb + bk);
    v16h b1 = *(const v16h*)(Z + (size_t)(bc + 16) * DIM + kb + bk);
    c00 = __builtin_amdgcn_wmma_f32_16x16x32_f16(false, a0, false, b0, (short)0, c00, false, false);
    c01 = __builtin_amdgcn_wmma_f32_16x16x32_f16(false, a0, false, b1, (short)0, c01, false, false);
    c10 = __builtin_amdgcn_wmma_f32_16x16x32_f16(false, a1, false, b0, (short)0, c10, false, false);
    c11 = __builtin_amdgcn_wmma_f32_16x16x32_f16(false, a1, false, b1, (short)0, c11, false, false);
  }
  // C/D layout: VGPR r -> M = r (lanes 0-15) / M = 8+r (lanes 16-31); N = lane&15
  const int mrow = i0 + (lane >> 4) * 8;
  const int mcol = j0 + (lane & 15);
  #pragma unroll
  for (int r = 0; r < 8; r++) {
    S[(size_t)(mrow + r) * NROWS + mcol]           = c00[r];
    S[(size_t)(mrow + r) * NROWS + mcol + 16]      = c01[r];
    S[(size_t)(mrow + 16 + r) * NROWS + mcol]      = c10[r];
    S[(size_t)(mrow + 16 + r) * NROWS + mcol + 16] = c11[r];
  }
}

// ---------------------------------------------------------------- per-row top-15 + softmax(v/T)
__global__ void __launch_bounds__(256) k_topk_softmax(const float* __restrict__ S,
                                                      int* __restrict__ oidx,
                                                      float* __restrict__ ow) {
  __shared__ float sv[NROWS];
  __shared__ float rv[256];
  __shared__ int   ri[256];
  __shared__ float selv[KNN];
  __shared__ int   seli[KNN];
  const int i = blockIdx.x, tid = threadIdx.x;
  const float* row = S + (size_t)i * NROWS;
  // async-stage the 16KB row into LDS (memory -> LDS, no VGPR round trip)
  #pragma unroll
  for (int j = tid * 4; j < NROWS; j += 1024) async_ld_lds_b128(&sv[j], row + j);
  wait_asynccnt0();
  __syncthreads();
  if (tid == 0) sv[i] = NEGINF;                       // mask diagonal
  __syncthreads();
  for (int r = 0; r < KNN; r++) {
    float bv = NEGINF; int bi = 0;
    for (int j = tid; j < NROWS; j += 256) {
      float v = sv[j];
      if (v > bv) { bv = v; bi = j; }
    }
    rv[tid] = bv; ri[tid] = bi;
    __syncthreads();
    for (int s = 128; s > 0; s >>= 1) {
      if (tid < s && rv[tid + s] > rv[tid]) { rv[tid] = rv[tid + s]; ri[tid] = ri[tid + s]; }
      __syncthreads();
    }
    if (tid == 0) { selv[r] = rv[0]; seli[r] = ri[0]; sv[ri[0]] = NEGINF; }
    __syncthreads();
  }
  if (tid == 0) {
    const float m = selv[0];
    float e[KNN], esum = 0.0f;
    for (int t = 0; t < KNN; t++) { e[t] = __expf((selv[t] - m) / TEMPR); esum += e[t]; }
    const float inv = 1.0f / esum;
    for (int t = 0; t < KNN; t++) {
      oidx[(size_t)i * KNN + t] = seli[t];
      ow[(size_t)i * KNN + t]   = e[t] * inv;
    }
  }
}

// ---------------------------------------------------------------- per-row top-32 of masked noise
__global__ void __launch_bounds__(256) k_neg_topk(const float* __restrict__ noise,
                                                  const int* __restrict__ idx_rna,
                                                  int* __restrict__ negidx) {
  __shared__ float sv[NROWS];
  __shared__ float rv[256];
  __shared__ int   ri[256];
  const int i = blockIdx.x, tid = threadIdx.x;
  const float* row = noise + (size_t)i * NROWS;
  #pragma unroll
  for (int j = tid * 4; j < NROWS; j += 1024) async_ld_lds_b128(&sv[j], row + j);
  wait_asynccnt0();
  __syncthreads();
  if (tid < KNN) sv[idx_rna[(size_t)i * KNN + tid]] = -2.0f;   // mask rna neighbors
  if (tid == KNN) sv[i] = -2.0f;                               // mask diagonal
  __syncthreads();
  for (int r = 0; r < NNEG; r++) {
    float bv = -3.0f; int bi = 0;
    for (int j = tid; j < NROWS; j += 256) {
      float v = sv[j];
      if (v > bv) { bv = v; bi = j; }
    }
    rv[tid] = bv; ri[tid] = bi;
    __syncthreads();
    for (int s = 128; s > 0; s >>= 1) {
      if (tid < s && rv[tid + s] > rv[tid]) { rv[tid] = rv[tid + s]; ri[tid] = ri[tid + s]; }
      __syncthreads();
    }
    if (tid == 0) { negidx[(size_t)i * NNEG + r] = ri[0]; sv[ri[0]] = -2.0f; }
    __syncthreads();
  }
}

// ---------------------------------------------------------------- align + attraction + repulsion + Laplacian
// one wave32 per row; lane holds dims (2*lane, 2*lane+1)
__global__ void __launch_bounds__(256) k_fused(const float* __restrict__ zraw,
                                               const float* __restrict__ zan,
                                               const int* __restrict__ idx_rna,
                                               const float* __restrict__ w_rna,
                                               const int* __restrict__ idx_atac,
                                               const float* __restrict__ w_atac,
                                               const int* __restrict__ negidx,
                                               float* __restrict__ acc) {
  const int lane = threadIdx.x & 31;
  const int warp = threadIdx.x >> 5;
  const int i = blockIdx.x * 8 + warp;
  const float2 zi = ((const float2*)(zraw + (size_t)i * DIM))[lane];
  const float2 ni = ((const float2*)(zan  + (size_t)i * DIM))[lane];

  float selfp = zi.x * zi.x + zi.y * zi.y;   // ||z_i||^2 partial
  float lapp = 0.0f;                         // Σ_j z_i·z_j partial (linear in dots)
  float attp = 0.0f;                         // Σ_j n_i·n_j partial
  for (int t = 0; t < KNN; t++) {
    const int j = idx_rna[(size_t)i * KNN + t];
    const float2 zj = ((const float2*)(zraw + (size_t)j * DIM))[lane];
    const float2 nj = ((const float2*)(zan  + (size_t)j * DIM))[lane];
    lapp += zi.x * zj.x + zi.y * zj.y;
    attp += ni.x * nj.x + ni.y * nj.y;
  }
  #pragma unroll
  for (int off = 16; off > 0; off >>= 1) {
    selfp += __shfl_down(selfp, off, 32);
    lapp  += __shfl_down(lapp,  off, 32);
    attp  += __shfl_down(attp,  off, 32);
  }

  // repulsion: relu(MARGIN - (1 - dot)) = relu(dot - (1-MARGIN)), needs full dot per sample
  float repsum = 0.0f;
  for (int r = 0; r < NNEG; r++) {
    const int j = negidx[(size_t)i * NNEG + r];
    const float2 nj = ((const float2*)(zan + (size_t)j * DIM))[lane];
    float d = ni.x * nj.x + ni.y * nj.y;
    #pragma unroll
    for (int off = 16; off > 0; off >>= 1) d += __shfl_down(d, off, 32);
    if (lane == 0) repsum += fmaxf(d - (1.0f - MARGIN), 0.0f);
  }

  // KL alignment on the RNA support: lane t<15 handles neighbor t
  float alp = 0.0f;
  if (lane < KNN) {
    const float tv = w_rna[(size_t)i * KNN + lane];
    const int   j  = idx_rna[(size_t)i * KNN + lane];
    float av = 0.0f;
    for (int u = 0; u < KNN; u++)
      if (idx_atac[(size_t)i * KNN + u] == j) av = w_atac[(size_t)i * KNN + u];
    alp = tv * (__logf(tv) - __logf(av + 1e-8f));
  }
  #pragma unroll
  for (int off = 16; off > 0; off >>= 1) alp += __shfl_down(alp, off, 32);

  if (lane == 0) {
    const float lap_i  = selfp - lapp * (1.0f / 15.0f);  // deg == 15 exactly
    const float attr_i = 1.0f - attp * (1.0f / 15.0f);
    atomicAdd(&acc[0], alp);
    atomicAdd(&acc[1], attr_i);
    atomicAdd(&acc[2], repsum);
    atomicAdd(&acc[3], lap_i);
  }
}

// ---------------------------------------------------------------- PPR diffusion + diff term, per row
// S5[i,:] = 0.2*Σ_{t=1..5} 0.8^{t-1} (A^t)[i,:] + 0.8^5 (A^6)[i,:]
// computed via v <- v·A (scatter of sparse rows) entirely in LDS.
__global__ void __launch_bounds__(256) k_ppr_diff(const int* __restrict__ idx_rna,
                                                  const float* __restrict__ w_rna,
                                                  const int* __restrict__ idx_atac,
                                                  const float* __restrict__ w_atac,
                                                  float* __restrict__ acc) {
  __shared__ float buf0[NROWS];
  __shared__ float buf1[NROWS];
  __shared__ float sacc[NROWS];
  __shared__ float red[256];
  const int i = blockIdx.x, tid = threadIdx.x;
  float* v = buf0;
  float* w = buf1;
  for (int j = tid; j < NROWS; j += 256) { v[j] = 0.0f; sacc[j] = 0.0f; }
  __syncthreads();
  if (tid < KNN) v[idx_rna[(size_t)i * KNN + tid]] = w_rna[(size_t)i * KNN + tid];
  __syncthreads();
  const float coef[6] = {0.2f, 0.16f, 0.128f, 0.1024f, 0.08192f, 0.32768f};
  for (int j = tid; j < NROWS; j += 256) sacc[j] += coef[0] * v[j];
  __syncthreads();
  for (int t = 1; t < 6; t++) {
    for (int j = tid; j < NROWS; j += 256) w[j] = 0.0f;
    __syncthreads();
    for (int k = tid; k < NROWS; k += 256) {
      const float vk = v[k];
      if (vk != 0.0f) {
        const int*   ip = idx_rna + (size_t)k * KNN;
        const float* wp = w_rna  + (size_t)k * KNN;
        for (int u = 0; u < KNN; u++) atomicAdd(&w[ip[u]], vk * wp[u]);  // ds_add_f32
      }
    }
    __syncthreads();
    const float c = coef[t];
    for (int j = tid; j < NROWS; j += 256) sacc[j] += c * w[j];
    __syncthreads();
    float* tmp = v; v = w; w = tmp;
  }
  // row sum and sum of squares of the (pre-normalized) S row
  float ps = 0.0f, ps2 = 0.0f;
  for (int j = tid; j < NROWS; j += 256) { const float x = sacc[j]; ps += x; ps2 += x * x; }
  red[tid] = ps; __syncthreads();
  for (int s = 128; s > 0; s >>= 1) { if (tid < s) red[tid] += red[tid + s]; __syncthreads(); }
  const float rowsum = red[0]; __syncthreads();
  red[tid] = ps2; __syncthreads();
  for (int s = 128; s > 0; s >>= 1) { if (tid < s) red[tid] += red[tid + s]; __syncthreads(); }
  const float sumsq = red[0]; __syncthreads();
  if (tid == 0) {
    const float inv = 1.0f / fmaxf(rowsum, 1e-8f);
    float cross = 0.0f, asq = 0.0f;
    for (int u = 0; u < KNN; u++) {
      const float av = w_atac[(size_t)i * KNN + u];
      const int   aj = idx_atac[(size_t)i * KNN + u];
      cross += av * sacc[aj] * inv;
      asq   += av * av;
    }
    // Σ_j (Aa_ij - S_ij)^2 = Σ S^2 - 2 Σ Aa*S + Σ Aa^2   (Aa sparse)
    atomicAdd(&acc[4], sumsq * inv * inv - 2.0f * cross + asq);
  }
}

// ---------------------------------------------------------------- finalize
__global__ void k_finalize(const float* __restrict__ acc, float* __restrict__ out) {
  if (threadIdx.x == 0 && blockIdx.x == 0) {
    const float fN = (float)NROWS;
    const float align = acc[0] / fN;
    const float contr = acc[1] / fN + acc[2] / (fN * (float)NNEG);
    const float lap   = acc[3] / fN;
    const float diff  = acc[4] / (fN * fN);
    out[0] = 1.0f * align + 1.0f * contr + 0.5f * lap + 0.5f * diff;
  }
}

// ================================================================ launch
extern "C" void kernel_launch(void* const* d_in, const int* in_sizes, int n_in,
                              void* d_out, int out_size, void* d_ws, size_t ws_size,
                              hipStream_t stream) {
  (void)in_sizes; (void)n_in; (void)out_size; (void)ws_size;
  const float* zr    = (const float*)d_in[0];   // z_rna  [N,64] f32
  const float* za    = (const float*)d_in[1];   // z_atac [N,64] f32
  const float* noise = (const float*)d_in[2];   // noise  [N,N]  f32

  char* p = (char*)d_ws;
  float* simbuf = (float*)p;        p += (size_t)NROWS * NROWS * sizeof(float);   // 64 MB, reused
  _Float16* zrh = (_Float16*)p;     p += (size_t)NROWS * DIM * sizeof(_Float16);
  _Float16* zah = (_Float16*)p;     p += (size_t)NROWS * DIM * sizeof(_Float16);
  float* zan    = (float*)p;        p += (size_t)NROWS * DIM * sizeof(float);
  int*   idx_rna  = (int*)p;        p += (size_t)NROWS * KNN * sizeof(int);
  float* w_rna    = (float*)p;      p += (size_t)NROWS * KNN * sizeof(float);
  int*   idx_atac = (int*)p;        p += (size_t)NROWS * KNN * sizeof(int);
  float* w_atac   = (float*)p;      p += (size_t)NROWS * KNN * sizeof(float);
  int*   negidx   = (int*)p;        p += (size_t)NROWS * NNEG * sizeof(int);
  float* acc      = (float*)p;

  k_init<<<1, 32, 0, stream>>>(acc);
  k_normalize<<<(2 * NROWS) / 8, 256, 0, stream>>>(zr, za, zrh, zah, zan);

  dim3 gg(NROWS / 64, NROWS / 128);
  k_sim_gemm<<<gg, 256, 0, stream>>>(zrh, simbuf);
  k_topk_softmax<<<NROWS, 256, 0, stream>>>(simbuf, idx_rna, w_rna);
  k_sim_gemm<<<gg, 256, 0, stream>>>(zah, simbuf);
  k_topk_softmax<<<NROWS, 256, 0, stream>>>(simbuf, idx_atac, w_atac);

  k_neg_topk<<<NROWS, 256, 0, stream>>>(noise, idx_rna, negidx);
  k_fused<<<NROWS / 8, 256, 0, stream>>>(za, zan, idx_rna, w_rna, idx_atac, w_atac, negidx, acc);
  k_ppr_diff<<<NROWS, 256, 0, stream>>>(idx_rna, w_rna, idx_atac, w_atac, acc);
  k_finalize<<<1, 1, 0, stream>>>(acc, (float*)d_out);
}
